// LocalizedPromptedAttentionLayer_46110768890010
// MI455X (gfx1250) — compile-verified, matrix operand order checked by reading, not run on hardware
//
#include <hip/hip_runtime.h>

// ---------------------------------------------------------------------------
// MI455X (gfx1250) implementation. wave32, WMMA bf16 16x16x32, f32 accum.
//
// Fused formulation:
//   q_x = x @ (Wq*Wx)^T + (Wq*bx + bq)          (never materialize x_proj)
//   k   = y @ (Wk*Wy)^T + (Wk*by + bk)          (never materialize y_proj)
//   v   = y @ (Wv*Wy)^T + (Wv*by + bv)
//   q_prompt = prompt @ Wq^T + bq               (shared by all (b,n))
//   attention per (b,n): [5x16] window, 12 heads, Dh=64
//   out = o @ out_w^T + out_b
// ---------------------------------------------------------------------------

typedef __attribute__((ext_vector_type(16))) __bf16 v16bf;
typedef __attribute__((ext_vector_type(8)))  float  v8f;

#define LP 40   // LDS row pitch (bf16) for GEMM tiles: 80B rows, conflict-free

// ---------------- constants (problem sizes from the reference) -------------
#define Bc   32
#define Nc   576
#define Wc   16
#define Dc   768
#define Pc   4
#define Hc   12
#define Dh   64
#define BN   (Bc * Nc)          // 18432
#define BNW  (BN * Wc)          // 294912
#define BNQ  (BN * (1 + Pc))    // 92160

// ---------------------------------------------------------------------------
// Tile staging: 128 rows x 32 K, f32 or bf16 source -> bf16 LDS (pitch LP)
// ---------------------------------------------------------------------------
__device__ __forceinline__ void stage_tile(const float* __restrict__ src, int ld,
                                           int row0, int k0, __bf16* dst, int tid) {
#pragma unroll
  for (int j = 0; j < 4; ++j) {
    int f = tid + j * 256;                 // 1024 float4 chunks
    int r = f >> 3;
    int c4 = (f & 7) * 4;
    const float4 v = *(const float4*)(src + (size_t)(row0 + r) * ld + k0 + c4);
    __bf16* p = dst + r * LP + c4;
    p[0] = (__bf16)v.x; p[1] = (__bf16)v.y; p[2] = (__bf16)v.z; p[3] = (__bf16)v.w;
    // speculative prefetch of next K tile (dropped if OOB)
    __builtin_prefetch(src + (size_t)(row0 + r) * ld + k0 + 32 + c4, 0, 1);
  }
}

__device__ __forceinline__ void stage_tile(const __bf16* __restrict__ src, int ld,
                                           int row0, int k0, __bf16* dst, int tid) {
#pragma unroll
  for (int j = 0; j < 2; ++j) {
    int g = tid + j * 256;                 // 512 chunks of 8 bf16 (16B)
    int r = g >> 2;
    int c8 = (g & 3) * 8;
    __builtin_memcpy(dst + r * LP + c8, src + (size_t)(row0 + r) * ld + k0 + c8, 16);
  }
}

// ---------------------------------------------------------------------------
// Workhorse: C[M,N] = A[M,K] @ B[N,K]^T (+bias). A: f32 or bf16, B: f32.
// Block 256 thr = 8 waves; tile 128x128; wave tile 64x32 = 4x2 WMMA frags.
// Requires M%128==0, N%128==0, K%32==0 (true for all shapes used here).
// ---------------------------------------------------------------------------
template <typename TA, typename TC>
__global__ __launch_bounds__(256) void gemm_xwt(
    const TA* __restrict__ A, const float* __restrict__ B,
    const float* __restrict__ bias, TC* __restrict__ C,
    int M, int N, int K) {
  __shared__ __bf16 As[128 * LP];
  __shared__ __bf16 Bs[128 * LP];
  const int tid  = threadIdx.x;
  const int lane = tid & 31;
  const int wave = tid >> 5;
  const int l16  = lane & 15;
  const int half = lane >> 4;
  const int m0   = blockIdx.x * 128;
  const int n0   = blockIdx.y * 128;
  const int wm0  = (wave & 1) * 64;
  const int wn0  = (wave >> 1) * 32;

  v8f acc[4][2];
#pragma unroll
  for (int i = 0; i < 4; ++i)
#pragma unroll
    for (int j = 0; j < 2; ++j)
#pragma unroll
      for (int e = 0; e < 8; ++e) acc[i][j][e] = 0.0f;

  for (int k0 = 0; k0 < K; k0 += 32) {
    stage_tile(A, K, m0, k0, As, tid);
    stage_tile(B, K, n0, k0, Bs, tid);
    __syncthreads();

    v16bf fa[4], fb[2];
#pragma unroll
    for (int i = 0; i < 4; ++i)
      __builtin_memcpy(&fa[i], &As[(wm0 + i * 16 + l16) * LP + half * 16], 32);
#pragma unroll
    for (int j = 0; j < 2; ++j)
      __builtin_memcpy(&fb[j], &Bs[(wn0 + j * 16 + l16) * LP + half * 16], 32);

#pragma unroll
    for (int i = 0; i < 4; ++i)
#pragma unroll
      for (int j = 0; j < 2; ++j)
        acc[i][j] = __builtin_amdgcn_wmma_f32_16x16x32_bf16(
            false, fa[i], false, fb[j], (short)0, acc[i][j], false, false);
    __syncthreads();
  }

#pragma unroll
  for (int j = 0; j < 2; ++j) {
    const int n = n0 + wn0 + j * 16 + l16;
    const float bv = bias ? bias[n] : 0.0f;
#pragma unroll
    for (int i = 0; i < 4; ++i) {
      const int mBase = m0 + wm0 + i * 16 + half * 8;
#pragma unroll
      for (int e = 0; e < 8; ++e) {
        float v = acc[i][j][e] + bv;
        C[(size_t)(mBase + e) * N + n] = (TC)v;
      }
    }
  }
}

// ---------------------------------------------------------------------------
// 32x32 LDS transpose (f32), out[c*R + r] = in[r*C + c]
// ---------------------------------------------------------------------------
__global__ __launch_bounds__(256) void transpose_f32(
    const float* __restrict__ in, float* __restrict__ out, int R, int C) {
  __shared__ float t[32][33];
  const int bx = blockIdx.x * 32, by = blockIdx.y * 32;
  const int x = threadIdx.x, y = threadIdx.y;   // blockDim (32, 8)
#pragma unroll
  for (int j = 0; j < 32; j += 8)
    t[y + j][x] = in[(size_t)(by + y + j) * C + bx + x];
  __syncthreads();
#pragma unroll
  for (int j = 0; j < 32; j += 8)
    out[(size_t)(bx + y + j) * R + by + x] = t[x][y + j];
}

// ---------------------------------------------------------------------------
// Fused biases + prompt queries (tiny):
//   bqx = Wq@bx + bq ; byk = Wk@by + bk ; byv = Wv@by + bv
//   qp[p,n] = sum_k prompt[p,k]*Wq[n,k] + bq[n]   (bf16)
// ---------------------------------------------------------------------------
__global__ void prep_small(const float* __restrict__ ipw, const float* __restrict__ ipb,
                           const float* __restrict__ bx, const float* __restrict__ by,
                           const float* __restrict__ prompt,
                           float* __restrict__ bqx, float* __restrict__ byk,
                           float* __restrict__ byv, __bf16* __restrict__ qp) {
  const int i = blockIdx.x * 256 + threadIdx.x;
  if (i < 768) {
    float s = ipb[i];
    for (int k = 0; k < 768; ++k) s += ipw[(size_t)i * 768 + k] * bx[k];
    bqx[i] = s;
  } else if (i < 1536) {
    const int n = i - 768;
    float s = ipb[768 + n];
    for (int k = 0; k < 768; ++k) s += ipw[(size_t)(768 + n) * 768 + k] * by[k];
    byk[n] = s;
  } else if (i < 2304) {
    const int n = i - 1536;
    float s = ipb[1536 + n];
    for (int k = 0; k < 768; ++k) s += ipw[(size_t)(1536 + n) * 768 + k] * by[k];
    byv[n] = s;
  } else if (i < 2304 + Pc * 768) {
    const int t = i - 2304;
    const int p = t / 768, n = t % 768;
    float s = ipb[n];
    for (int k = 0; k < 768; ++k) s += prompt[(size_t)p * 768 + k] * ipw[(size_t)n * 768 + k];
    qp[(size_t)p * 768 + n] = (__bf16)s;
  }
}

// ---------------------------------------------------------------------------
// Attention: one block per (b,n). 128 thr = 4 waves; wave handles heads
// {w, w+4, w+8}. Scores: 2x wmma bf16 (M padded 5->16). Softmax over 16 keys
// via shfl within 16-lane halves. o = attn@v: wmma with K padded 16->32
// (upper-half-lane fragments zeroed), v staged transposed in LDS.
// ---------------------------------------------------------------------------
__global__ __launch_bounds__(128) void attn_kernel(
    const __bf16* __restrict__ Q,   // [BN, 768]
    const __bf16* __restrict__ QP,  // [4, 768]
    const __bf16* __restrict__ Km,  // [BN*16, 768]
    const __bf16* __restrict__ Vm,  // [BN*16, 768]
    __bf16* __restrict__ O)         // [BN*5, 768]
{
  __shared__ __bf16 vt[768 * 24];       // vt[d][kk], pitch 24 (48B, 16B aligned)
  __shared__ __bf16 as_[4][16 * 24];    // per-wave attn scratch [m][n]

  const int bn   = blockIdx.x;
  const int tid  = threadIdx.x;
  const int lane = tid & 31;
  const int wave = tid >> 5;
  const int l16  = lane & 15;
  const int half = lane >> 4;

  // stage v transposed (coalesced global reads)
  for (int r = 0; r < 16; ++r)
    for (int c = tid; c < 768; c += 128)
      vt[c * 24 + r] = Vm[((size_t)bn * 16 + r) * 768 + c];
  __syncthreads();

  // A-fragment rows for scores: row 0 = x-query, 1..4 = prompts, >=5 harmless
  const __bf16* qrow = (l16 == 0) ? (Q + (size_t)bn * 768)
                     : (l16 < 5)  ? (QP + (size_t)(l16 - 1) * 768)
                                  : QP;   // finite garbage, never stored
  const __bf16* krow = Km + ((size_t)bn * 16 + l16) * 768;
  __bf16* asw = &as_[wave][0];

  for (int hi = 0; hi < 3; ++hi) {
    const int h  = wave + 4 * hi;
    const int d0 = h * Dh;

    // ---- scores = q_h @ k_h^T  (16x16, K=64) ----
    v8f sc;
#pragma unroll
    for (int e = 0; e < 8; ++e) sc[e] = 0.0f;
#pragma unroll
    for (int kk = 0; kk < 64; kk += 32) {
      v16bf fq, fk;
      __builtin_memcpy(&fq, qrow + d0 + kk + half * 16, 32);
      __builtin_memcpy(&fk, krow + d0 + kk + half * 16, 32);
      sc = __builtin_amdgcn_wmma_f32_16x16x32_bf16(
          false, fq, false, fk, (short)0, sc, false, false);
    }

    // ---- softmax over the 16 keys (row m = e + 8*half, key = l16) ----
#pragma unroll
    for (int e = 0; e < 8; ++e) {
      float s  = sc[e] * 0.125f;   // 1/sqrt(64)
      float mx = s;
      mx = fmaxf(mx, __shfl_xor(mx, 1, 32));
      mx = fmaxf(mx, __shfl_xor(mx, 2, 32));
      mx = fmaxf(mx, __shfl_xor(mx, 4, 32));
      mx = fmaxf(mx, __shfl_xor(mx, 8, 32));
      float p   = __expf(s - mx);
      float sum = p;
      sum += __shfl_xor(sum, 1, 32);
      sum += __shfl_xor(sum, 2, 32);
      sum += __shfl_xor(sum, 4, 32);
      sum += __shfl_xor(sum, 8, 32);
      asw[(e + 8 * half) * 24 + l16] = (__bf16)(p / sum);
    }
    __syncthreads();

    // ---- o = attn @ v_h  (K padded 16->32: upper-half lanes carry zeros) ----
    v16bf fa;
    if (half == 0) {
      __builtin_memcpy(&fa, &asw[l16 * 24], 32);
    } else {
#pragma unroll
      for (int e = 0; e < 16; ++e) fa[e] = (__bf16)0.0f;
    }
#pragma unroll
    for (int nt = 0; nt < 4; ++nt) {
      v16bf fv;
      if (half == 0) {
        __builtin_memcpy(&fv, &vt[(d0 + nt * 16 + l16) * 24], 32);
      } else {
#pragma unroll
        for (int e = 0; e < 16; ++e) fv[e] = (__bf16)0.0f;
      }
      v8f oc;
#pragma unroll
      for (int e = 0; e < 8; ++e) oc[e] = 0.0f;
      oc = __builtin_amdgcn_wmma_f32_16x16x32_bf16(
          false, fa, false, fv, (short)0, oc, false, false);
      if (half == 0) {
#pragma unroll
        for (int e = 0; e < 5; ++e)
          O[((size_t)bn * 5 + e) * 768 + d0 + nt * 16 + l16] = (__bf16)oc[e];
      }
    }
    __syncthreads();
  }
}

// ---------------------------------------------------------------------------
extern "C" void kernel_launch(void* const* d_in, const int* in_sizes, int n_in,
                              void* d_out, int out_size, void* d_ws, size_t ws_size,
                              hipStream_t stream) {
  (void)in_sizes; (void)n_in; (void)out_size; (void)ws_size;
  const float* x      = (const float*)d_in[0];   // [BN, 1536]
  const float* y      = (const float*)d_in[1];   // [BNW, 1024]
  const float* Wx     = (const float*)d_in[2];   // [768, 1536]
  const float* bx     = (const float*)d_in[3];
  const float* Wy     = (const float*)d_in[4];   // [768, 1024]
  const float* by     = (const float*)d_in[5];
  const float* prompt = (const float*)d_in[6];   // [4, 768]
  const float* ipw    = (const float*)d_in[7];   // [2304, 768]
  const float* ipb    = (const float*)d_in[8];
  const float* ow     = (const float*)d_in[9];   // [768, 768]
  const float* ob     = (const float*)d_in[10];
  float* out = (float*)d_out;                    // [BNQ, 768]

  char* p = (char*)d_ws;
  auto take = [&](size_t bytes) { char* r = p; p += (bytes + 255) & ~(size_t)255; return r; };
  float*  WxT = (float*)take((size_t)1536 * 768 * 4);
  float*  WyT = (float*)take((size_t)1024 * 768 * 4);
  float*  Wqx = (float*)take((size_t)768 * 1536 * 4);
  float*  Wyk = (float*)take((size_t)768 * 1024 * 4);
  float*  Wyv = (float*)take((size_t)768 * 1024 * 4);
  float*  bqx = (float*)take(768 * 4);
  float*  byk = (float*)take(768 * 4);
  float*  byv = (float*)take(768 * 4);
  __bf16* qp  = (__bf16*)take((size_t)Pc * 768 * 2);
  __bf16* qx  = (__bf16*)take((size_t)BN * 768 * 2);
  __bf16* kb  = (__bf16*)take((size_t)BNW * 768 * 2);
  __bf16* vb  = (__bf16*)take((size_t)BNW * 768 * 2);
  __bf16* ob_ = (__bf16*)take((size_t)BNQ * 768 * 2);

  // 1) transposes so weight fusion is in X@W^T form
  transpose_f32<<<dim3(1536 / 32, 768 / 32), dim3(32, 8), 0, stream>>>(Wx, WxT, 768, 1536);
  transpose_f32<<<dim3(1024 / 32, 768 / 32), dim3(32, 8), 0, stream>>>(Wy, WyT, 768, 1024);
  // 2) fused biases + prompt queries
  prep_small<<<(2304 + Pc * 768 + 255) / 256, 256, 0, stream>>>(
      ipw, ipb, bx, by, prompt, bqx, byk, byv, qp);
  // 3) fused weights: Wqx = Wq@Wx, Wyk = Wk@Wy, Wyv = Wv@Wy
  gemm_xwt<float, float><<<dim3(768 / 128, 1536 / 128), 256, 0, stream>>>(
      ipw, WxT, nullptr, Wqx, 768, 1536, 768);
  gemm_xwt<float, float><<<dim3(768 / 128, 1024 / 128), 256, 0, stream>>>(
      ipw + (size_t)768 * 768, WyT, nullptr, Wyk, 768, 1024, 768);
  gemm_xwt<float, float><<<dim3(768 / 128, 1024 / 128), 256, 0, stream>>>(
      ipw + (size_t)1536 * 768, WyT, nullptr, Wyv, 768, 1024, 768);
  // 4) q/k/v projections (bf16 outputs)
  gemm_xwt<float, __bf16><<<dim3(BN / 128, 768 / 128), 256, 0, stream>>>(
      x, Wqx, bqx, qx, BN, 768, 1536);
  gemm_xwt<float, __bf16><<<dim3(BNW / 128, 768 / 128), 256, 0, stream>>>(
      y, Wyk, byk, kb, BNW, 768, 1024);
  gemm_xwt<float, __bf16><<<dim3(BNW / 128, 768 / 128), 256, 0, stream>>>(
      y, Wyv, byv, vb, BNW, 768, 1024);
  // 5) attention
  attn_kernel<<<BN, 128, 0, stream>>>(qx, qp, kb, vb, ob_);
  // 6) output projection -> d_out (f32, +bias)
  gemm_xwt<__bf16, float><<<dim3(BNQ / 128, 768 / 128), 256, 0, stream>>>(
      ob_, ow, ob, out, BNQ, 768, 768);
}
